// MultiheadSelfAttention_37795712205561
// MI455X (gfx1250) — matrix operand order repaired and checked by reading, not compiled, not run on hardware
//
#include <hip/hip_runtime.h>
#include <hip/hip_bf16.h>
#include <stdint.h>

#define D_MODEL 1024
#define NUM_HEADS 16
#define D_K 64
#define SEQ 2048
#define BATCH 2

// Block-GEMM tile config: 128x64 output tile, K-step 32, 8 waves (4M x 2N),
// each wave a 32x32 quad of 16x16 WMMA tiles.
#define BLK_M 128
#define BLK_N 64
#define BLK_K 32
#define A_BYTES (BLK_M * BLK_K * 2)               // 8 KB
#define B_BYTES (BLK_N * BLK_K * 2)               // 4 KB
#define STAGE_BYTES (A_BYTES + B_BYTES)           // 12 KB (x2 stages = 24 KB LDS)
#define STAGE_ELEMS (STAGE_BYTES / 2)

typedef __bf16 bf16;
typedef __attribute__((ext_vector_type(16))) __bf16 v16bf;
typedef __attribute__((ext_vector_type(8)))  float  v8f;

// ---------------------------------------------------------------------------
// WMMA wrapper: D(16x16,f32) = A(16x32,bf16) * B(32x16,bf16) + C
// ---------------------------------------------------------------------------
__device__ __forceinline__ v8f wmma_bf16(v16bf a, v16bf b, v8f c) {
    return __builtin_amdgcn_wmma_f32_16x16x32_bf16(
        /*neg_a=*/false, a, /*neg_b=*/false, b,
        /*c_mod=*/(short)0, c, /*reuse_a=*/false, /*reuse_b=*/false);
}

// ---------------------------------------------------------------------------
// CDNA5 async copy: per-lane global -> LDS, tracked by ASYNCcnt.
// lds_off: wave-relative LDS byte address (single __shared__ array per kernel
// => array base is LDS offset 0). lds_dst: the generic pointer to the same
// destination — passed as an operand so the LDS buffer ESCAPES into the asm;
// otherwise alias analysis proves the asm can't write smem and deletes the
// post-barrier ds_load fragment reads (observed in round 2 disasm).
// ---------------------------------------------------------------------------
__device__ __forceinline__ void async_load_b128(uint32_t lds_off, const bf16* gaddr,
                                                void* lds_dst) {
    asm volatile("global_load_async_to_lds_b128 %0, %1, off"
                 :: "v"(lds_off), "v"((uint64_t)(uintptr_t)gaddr), "v"(lds_dst)
                 : "memory");
}
__device__ __forceinline__ void wait_async0() {
    asm volatile("s_wait_asynccnt 0" ::: "memory");
}

// ---------------------------------------------------------------------------
// A-fragment (16x32 bf16, row-major source, ld in elements).
// Lanes 0-15 hold M=lane, K=0..15; lanes 16-31 hold K=16..31 (matches B).
// Each VGPR packs (K, K+1); dwords 0-7 are one contiguous 32B run.
// ---------------------------------------------------------------------------
__device__ __forceinline__ v16bf load_A_frag(const bf16* base, int ld) {
    const int lane = threadIdx.x & 31;
    const int m    = lane & 15;
    const int hi   = lane >> 4;
    union { v16bf v; uint32_t u[8]; } f;
    const uint32_t* rowp = (const uint32_t*)(base + (size_t)m * ld);
#pragma unroll
    for (int v = 0; v < 8; ++v) {
        const int k = 2 * v + hi * 16;
        f.u[v] = rowp[k >> 1];
    }
    return f.v;
}

// ---------------------------------------------------------------------------
// B-fragment (32x16 bf16). Source "N-major": element (k,n) at base[n*ld + k].
// Lanes 0-15: N=lane, K=0..15; lanes 16-31: N=lane-16, K=16..31.
// ---------------------------------------------------------------------------
__device__ __forceinline__ v16bf load_B_frag(const bf16* base, int ld) {
    const int lane = threadIdx.x & 31;
    const int n    = lane & 15;
    const int hi   = lane >> 4;
    union { v16bf v; uint32_t u[8]; } f;
    const uint32_t* colp = (const uint32_t*)(base + (size_t)n * ld + hi * 16);
#pragma unroll
    for (int v = 0; v < 8; ++v) f.u[v] = colp[v];
    return f.v;
}

// Cross-lane reductions confined to a 16-lane half (C-layout row groups).
__device__ __forceinline__ float half_max(float v) {
    v = fmaxf(v, __shfl_xor(v, 1, 32));
    v = fmaxf(v, __shfl_xor(v, 2, 32));
    v = fmaxf(v, __shfl_xor(v, 4, 32));
    v = fmaxf(v, __shfl_xor(v, 8, 32));
    return v;
}
__device__ __forceinline__ float half_sum(float v) {
    v += __shfl_xor(v, 1, 32);
    v += __shfl_xor(v, 2, 32);
    v += __shfl_xor(v, 4, 32);
    v += __shfl_xor(v, 8, 32);
    return v;
}

// ---------------------------------------------------------------------------
// fp32 -> bf16 cast
// ---------------------------------------------------------------------------
__global__ void cast_f32_bf16_kernel(const float* __restrict__ src,
                                     bf16* __restrict__ dst, int n) {
    int i = blockIdx.x * blockDim.x + threadIdx.x;
    if (i < n) dst[i] = (bf16)src[i];
}

// ---------------------------------------------------------------------------
// Async stagers for the block GEMM.
// A tile: BLK_M x BLK_K row-major (ld=BLK_K) = 8KB -> 512 x b128, 2/thread.
// B tile: BLK_N x BLK_K n-major  (ld=BLK_K) = 4KB -> 256 x b128, 1/thread.
// ---------------------------------------------------------------------------
__device__ __forceinline__ void issue_tileA(const bf16* A, int tid,
                                            uint32_t lds_base, bf16* smem) {
#pragma unroll
    for (int i = 0; i < 2; ++i) {
        const int idx = tid + i * 256;
        const int row = idx >> 2;        // 0..127
        const int c16 = idx & 3;         // 16B chunk within 64B row
        const uint32_t off = lds_base + row * (BLK_K * 2) + c16 * 16;
        async_load_b128(off, A + (size_t)row * D_MODEL + c16 * 8,
                        (char*)smem + off);
    }
}
__device__ __forceinline__ void issue_tileB(const bf16* Bm, int tid,
                                            uint32_t lds_base, bf16* smem) {
    const int row = tid >> 2;            // n = 0..63
    const int c16 = tid & 3;
    const uint32_t off = lds_base + row * (BLK_K * 2) + c16 * 16;
    async_load_b128(off, Bm + (size_t)row * D_MODEL + c16 * 8,
                    (char*)smem + off);
}

// ---------------------------------------------------------------------------
// QKV projection GEMM: Y[R,C] = sum_d X[R,d] * W[C,d]   (y = x @ W^T)
// mode 0: Q -> (B,H,S,dk) scaled 1/sqrt(dk);  mode 1: K -> (B,H,S,dk)
// mode 2: V -> (B,H,dk,S) transposed.
// LDS double-buffered via global_load_async_to_lds_b128.
// ---------------------------------------------------------------------------
__device__ __forceinline__ void store_tile_qkv(bf16* __restrict__ Out, v8f acc,
                                               int rowBase, int colBase,
                                               int lane, int mode, float qscale) {
#pragma unroll
    for (int r = 0; r < 8; ++r) {
        const int m  = (lane < 16) ? r : (r + 8);
        const int n  = lane & 15;
        const int R  = rowBase + m;
        const int C  = colBase + n;
        const int b_ = R / SEQ, s = R % SEQ;
        const int h  = C / D_K,  d = C % D_K;
        const float val = acc[r] * qscale;
        size_t addr;
        if (mode == 2)
            addr = ((size_t)(b_ * NUM_HEADS + h) * D_K + d) * SEQ + s;
        else
            addr = ((size_t)(b_ * NUM_HEADS + h) * SEQ + s) * D_K + d;
        Out[addr] = (bf16)val;
    }
}

__global__ __launch_bounds__(256) void gemm_qkv_kernel(
    const bf16* __restrict__ X, const bf16* __restrict__ W,
    bf16* __restrict__ Out, int mode) {
    __shared__ bf16 smem[2 * STAGE_ELEMS];   // 24 KB, LDS offset 0

    const int tid  = threadIdx.x;
    const int lane = tid & 31;
    const int wave = tid >> 5;
    const int mw   = wave & 3;               // 4 wave-rows of 32
    const int nw   = wave >> 2;              // 2 wave-cols of 32
    const int blockM = blockIdx.x * BLK_M;
    const int blockN = blockIdx.y * BLK_N;

    const bf16* Ab = X + (size_t)blockM * D_MODEL;
    const bf16* Bb = W + (size_t)blockN * D_MODEL;

    // Prologue: stage k0 = 0 into stage 0.
    issue_tileA(Ab, tid, 0, smem);
    issue_tileB(Bb, tid, A_BYTES, smem);

    v8f acc00 = {}, acc01 = {}, acc10 = {}, acc11 = {};
    int stage = 0;
    for (int k0 = 0; k0 < D_MODEL; k0 += BLK_K) {
        wait_async0();        // our stage's async copies landed in LDS
        __syncthreads();      // ... for every wave in the block

        const int nk = k0 + BLK_K;
        if (nk < D_MODEL) {   // prefetch next stage while we compute
            const uint32_t nb = (uint32_t)(stage ^ 1) * STAGE_BYTES;
            issue_tileA(Ab + nk, tid, nb, smem);
            issue_tileB(Bb + nk, tid, nb + A_BYTES, smem);
        }

        const bf16* sa = smem + (size_t)stage * STAGE_ELEMS;
        const bf16* sb = sa + BLK_M * BLK_K;
        v16bf a0 = load_A_frag(sa + (mw * 32 +  0) * BLK_K, BLK_K);
        v16bf a1 = load_A_frag(sa + (mw * 32 + 16) * BLK_K, BLK_K);
        v16bf b0 = load_B_frag(sb + (nw * 32 +  0) * BLK_K, BLK_K);
        v16bf b1 = load_B_frag(sb + (nw * 32 + 16) * BLK_K, BLK_K);
        acc00 = wmma_bf16(a0, b0, acc00);
        acc01 = wmma_bf16(a0, b1, acc01);
        acc10 = wmma_bf16(a1, b0, acc10);
        acc11 = wmma_bf16(a1, b1, acc11);

        __syncthreads();      // all reads done before this stage is overwritten
        stage ^= 1;
    }

    const float qscale = (mode == 0) ? 0.125f : 1.0f;   // 1/sqrt(64)
    const int rb = blockM + mw * 32;
    const int cb = blockN + nw * 32;
    store_tile_qkv(Out, acc00, rb +  0, cb +  0, lane, mode, qscale);
    store_tile_qkv(Out, acc01, rb +  0, cb + 16, lane, mode, qscale);
    store_tile_qkv(Out, acc10, rb + 16, cb +  0, lane, mode, qscale);
    store_tile_qkv(Out, acc11, rb + 16, cb + 16, lane, mode, qscale);
}

// ---------------------------------------------------------------------------
// RoPE, interleaved pairs, in-place on Q and K (B,H,S,dk) bf16.
// ---------------------------------------------------------------------------
__global__ void rope_kernel(bf16* __restrict__ Q, bf16* __restrict__ K, int npairs) {
    int idx = blockIdx.x * blockDim.x + threadIdx.x;
    if (idx >= npairs) return;
    const int p = idx & 31;
    const int s = (idx >> 5) % SEQ;
    const float inv = __expf(-(float)(2 * p) * 0.14391157f);  // ln(1e4)/64
    const float ang = (float)s * inv;
    const float cs = __cosf(ang), sn = __sinf(ang);
    const size_t off = (size_t)idx * 2;
    {
        float e = (float)Q[off], o = (float)Q[off + 1];
        Q[off]     = (bf16)(e * cs - o * sn);
        Q[off + 1] = (bf16)(e * sn + o * cs);
    }
    {
        float e = (float)K[off], o = (float)K[off + 1];
        K[off]     = (bf16)(e * cs - o * sn);
        K[off + 1] = (bf16)(e * sn + o * cs);
    }
}

// ---------------------------------------------------------------------------
// Flash attention: 1 wave = one 16-row Q tile; 32-key blocks; causal online
// softmax; P repacked C->A layout through per-wave LDS.
// ---------------------------------------------------------------------------
__global__ __launch_bounds__(128) void attn_kernel(
    const bf16* __restrict__ Q, const bf16* __restrict__ K,
    const bf16* __restrict__ Vt, bf16* __restrict__ AO) {
    __shared__ bf16 lds_p[4][16][32];

    const int lane  = threadIdx.x & 31;
    const int wave  = threadIdx.x >> 5;
    const int bh    = blockIdx.y;
    const int qTile = blockIdx.x * 4 + wave;

    const bf16* Qh = Q  + (size_t)bh * SEQ * D_K;
    const bf16* Kh = K  + (size_t)bh * SEQ * D_K;
    const bf16* Vh = Vt + (size_t)bh * D_K * SEQ;

    const v16bf aq0 = load_A_frag(Qh + (size_t)qTile * 16 * D_K + 0,  D_K);
    const v16bf aq1 = load_A_frag(Qh + (size_t)qTile * 16 * D_K + 32, D_K);

    v8f o0 = {}, o1 = {}, o2 = {}, o3 = {};
    float mrun[8], lrun[8];
#pragma unroll
    for (int r = 0; r < 8; ++r) { mrun[r] = -1e30f; lrun[r] = 0.0f; }

    const int nkb = (qTile * 16 + 16 + 31) / 32;
    for (int kb = 0; kb < nkb; ++kb) {
        const int key0 = kb * 32;
        __builtin_prefetch(Kh + (size_t)(key0 + 32) * D_K, 0, 1);
        __builtin_prefetch(Vh + key0 + 32, 0, 1);

        v8f s0 = {}, s1 = {};
        {
            v16bf b00 = load_B_frag(Kh + (size_t)(key0 + 0)  * D_K + 0,  D_K);
            v16bf b01 = load_B_frag(Kh + (size_t)(key0 + 0)  * D_K + 32, D_K);
            s0 = wmma_bf16(aq0, b00, s0);
            s0 = wmma_bf16(aq1, b01, s0);
            v16bf b10 = load_B_frag(Kh + (size_t)(key0 + 16) * D_K + 0,  D_K);
            v16bf b11 = load_B_frag(Kh + (size_t)(key0 + 16) * D_K + 32, D_K);
            s1 = wmma_bf16(aq0, b10, s1);
            s1 = wmma_bf16(aq1, b11, s1);
        }

        if (key0 + 31 > qTile * 16) {   // wave-uniform branch, VALU mask inside
#pragma unroll
            for (int r = 0; r < 8; ++r) {
                const int m   = (lane < 16) ? r : (r + 8);
                const int row = qTile * 16 + m;
                const int c0  = key0 + (lane & 15);
                if (c0 > row)      s0[r] = -1e30f;
                if (c0 + 16 > row) s1[r] = -1e30f;
            }
        }

        float p0[8], p1[8];
#pragma unroll
        for (int r = 0; r < 8; ++r) {
            float bm   = half_max(fmaxf(s0[r], s1[r]));
            float mnew = fmaxf(mrun[r], bm);
            float corr = __expf(mrun[r] - mnew);
            mrun[r] = mnew;
            float e0 = __expf(s0[r] - mnew);
            float e1 = __expf(s1[r] - mnew);
            p0[r] = e0; p1[r] = e1;
            lrun[r] = lrun[r] * corr + half_sum(e0 + e1);
            o0[r] *= corr; o1[r] *= corr; o2[r] *= corr; o3[r] *= corr;
        }

        // C-layout f32 -> LDS row-major bf16 -> A fragment (same-wave, in-order DS).
#pragma unroll
        for (int r = 0; r < 8; ++r) {
            const int m = (lane < 16) ? r : (r + 8);
            const int c = lane & 15;
            lds_p[wave][m][c]      = (bf16)p0[r];
            lds_p[wave][m][c + 16] = (bf16)p1[r];
        }
        v16bf ap = load_A_frag(&lds_p[wave][0][0], 32);

        v16bf bv0 = load_B_frag(Vh + (size_t)(0 * 16) * SEQ + key0, SEQ);
        v16bf bv1 = load_B_frag(Vh + (size_t)(1 * 16) * SEQ + key0, SEQ);
        v16bf bv2 = load_B_frag(Vh + (size_t)(2 * 16) * SEQ + key0, SEQ);
        v16bf bv3 = load_B_frag(Vh + (size_t)(3 * 16) * SEQ + key0, SEQ);
        o0 = wmma_bf16(ap, bv0, o0);
        o1 = wmma_bf16(ap, bv1, o1);
        o2 = wmma_bf16(ap, bv2, o2);
        o3 = wmma_bf16(ap, bv3, o3);
    }

    const int b_ = bh / NUM_HEADS, h = bh % NUM_HEADS;
#pragma unroll
    for (int r = 0; r < 8; ++r) {
        const int m = (lane < 16) ? r : (r + 8);
        const int n = lane & 15;
        const int s = qTile * 16 + m;
        const float inv_l = 1.0f / lrun[r];
        bf16* rowp = AO + ((size_t)(b_ * SEQ + s)) * D_MODEL + h * D_K;
        rowp[ 0 + n] = (bf16)(o0[r] * inv_l);
        rowp[16 + n] = (bf16)(o1[r] * inv_l);
        rowp[32 + n] = (bf16)(o2[r] * inv_l);
        rowp[48 + n] = (bf16)(o3[r] * inv_l);
    }
}

// ---------------------------------------------------------------------------
// Output projection: d_out[R,C] = sum_d AO[R,d] * Wo[C,d], fp32 output.
// Same LDS double-buffered block GEMM.
// ---------------------------------------------------------------------------
__global__ __launch_bounds__(256) void gemm_out_kernel(
    const bf16* __restrict__ X, const bf16* __restrict__ W,
    float* __restrict__ Y) {
    __shared__ bf16 smem[2 * STAGE_ELEMS];

    const int tid  = threadIdx.x;
    const int lane = tid & 31;
    const int wave = tid >> 5;
    const int mw   = wave & 3;
    const int nw   = wave >> 2;
    const int blockM = blockIdx.x * BLK_M;
    const int blockN = blockIdx.y * BLK_N;

    const bf16* Ab = X + (size_t)blockM * D_MODEL;
    const bf16* Bb = W + (size_t)blockN * D_MODEL;

    issue_tileA(Ab, tid, 0, smem);
    issue_tileB(Bb, tid, A_BYTES, smem);

    v8f acc00 = {}, acc01 = {}, acc10 = {}, acc11 = {};
    int stage = 0;
    for (int k0 = 0; k0 < D_MODEL; k0 += BLK_K) {
        wait_async0();
        __syncthreads();

        const int nk = k0 + BLK_K;
        if (nk < D_MODEL) {
            const uint32_t nb = (uint32_t)(stage ^ 1) * STAGE_BYTES;
            issue_tileA(Ab + nk, tid, nb, smem);
            issue_tileB(Bb + nk, tid, nb + A_BYTES, smem);
        }

        const bf16* sa = smem + (size_t)stage * STAGE_ELEMS;
        const bf16* sb = sa + BLK_M * BLK_K;
        v16bf a0 = load_A_frag(sa + (mw * 32 +  0) * BLK_K, BLK_K);
        v16bf a1 = load_A_frag(sa + (mw * 32 + 16) * BLK_K, BLK_K);
        v16bf b0 = load_B_frag(sb + (nw * 32 +  0) * BLK_K, BLK_K);
        v16bf b1 = load_B_frag(sb + (nw * 32 + 16) * BLK_K, BLK_K);
        acc00 = wmma_bf16(a0, b0, acc00);
        acc01 = wmma_bf16(a0, b1, acc01);
        acc10 = wmma_bf16(a1, b0, acc10);
        acc11 = wmma_bf16(a1, b1, acc11);

        __syncthreads();
        stage ^= 1;
    }

    const int rb = blockM + mw * 32;
    const int cb = blockN + nw * 32;
#pragma unroll
    for (int t = 0; t < 4; ++t) {
        const v8f acc = (t == 0) ? acc00 : (t == 1) ? acc01 : (t == 2) ? acc10 : acc11;
        const int rowBase = rb + (t >> 1) * 16;
        const int colBase = cb + (t & 1) * 16;
#pragma unroll
        for (int r = 0; r < 8; ++r) {
            const int m = (lane < 16) ? r : (r + 8);
            const int n = lane & 15;
            Y[(size_t)(rowBase + m) * D_MODEL + colBase + n] = acc[r];
        }
    }
}

// ---------------------------------------------------------------------------
// Host-side orchestration (graph-capture safe: launches only).
// ---------------------------------------------------------------------------
extern "C" void kernel_launch(void* const* d_in, const int* in_sizes, int n_in,
                              void* d_out, int out_size, void* d_ws, size_t ws_size,
                              hipStream_t stream) {
    const float* x  = (const float*)d_in[0];
    const float* wq = (const float*)d_in[1];
    const float* wk = (const float*)d_in[2];
    const float* wv = (const float*)d_in[3];
    const float* wo = (const float*)d_in[4];
    float* out = (float*)d_out;

    char* ws = (char*)d_ws;
    const size_t MB = 1u << 20;
    bf16* xb  = (bf16*)(ws + 0 * MB);    // (B*S, D)   bf16, 8 MB
    bf16* wqb = (bf16*)(ws + 8 * MB);    // (D, D)     bf16, 2 MB each
    bf16* wkb = (bf16*)(ws + 10 * MB);
    bf16* wvb = (bf16*)(ws + 12 * MB);
    bf16* wob = (bf16*)(ws + 14 * MB);
    bf16* qb  = (bf16*)(ws + 16 * MB);   // (B,H,S,dk)
    bf16* kb  = (bf16*)(ws + 24 * MB);   // (B,H,S,dk)
    bf16* vtb = (bf16*)(ws + 32 * MB);   // (B,H,dk,S)
    bf16* aob = (bf16*)(ws + 40 * MB);   // (B,S,D)   -> 48 MB total

    const int NX = BATCH * SEQ * D_MODEL;
    const int NW = D_MODEL * D_MODEL;

    cast_f32_bf16_kernel<<<(NX + 255) / 256, 256, 0, stream>>>(x,  xb,  NX);
    cast_f32_bf16_kernel<<<(NW + 255) / 256, 256, 0, stream>>>(wq, wqb, NW);
    cast_f32_bf16_kernel<<<(NW + 255) / 256, 256, 0, stream>>>(wk, wkb, NW);
    cast_f32_bf16_kernel<<<(NW + 255) / 256, 256, 0, stream>>>(wv, wvb, NW);
    cast_f32_bf16_kernel<<<(NW + 255) / 256, 256, 0, stream>>>(wo, wob, NW);

    // QKV projections: grid (M/128, N/64)
    dim3 ggrid(BATCH * SEQ / BLK_M, D_MODEL / BLK_N);
    gemm_qkv_kernel<<<ggrid, 256, 0, stream>>>(xb, wqb, qb,  0);
    gemm_qkv_kernel<<<ggrid, 256, 0, stream>>>(xb, wkb, kb,  1);
    gemm_qkv_kernel<<<ggrid, 256, 0, stream>>>(xb, wvb, vtb, 2);

    const int NPAIR = BATCH * NUM_HEADS * SEQ * (D_K / 2);
    rope_kernel<<<(NPAIR + 255) / 256, 256, 0, stream>>>(qb, kb, NPAIR);

    attn_kernel<<<dim3(SEQ / 64, BATCH * NUM_HEADS), 128, 0, stream>>>(qb, kb, vtb, aob);

    gemm_out_kernel<<<ggrid, 256, 0, stream>>>(aob, wob, out);
}
